// Decoder_22196390985918
// MI455X (gfx1250) — matrix operand order repaired and checked by reading, not compiled, not run on hardware
//
#include <hip/hip_runtime.h>
#include <hip/hip_bf16.h>

// ---------------------------------------------------------------------------
// SEAN/SPADE decoder block for MI455X (gfx1250, wave32, WMMA).
// - conv(middle) over the piecewise-constant 512-ch "middle" tensor is
//   replaced by a per-(batch,region,tap) table gather (kills 77 GFLOP).
// - SPADE 128->64 3x3 convs (gamma & beta stacked, M=128, K=1152) run as a
//   bf16 implicit GEMM on v_wmma_f32_16x16x32_bf16, N=32 pixels per wg.
// - Epilogue (instance-norm apply + table gather + sigmoid blend) is FUSED
//   into the GEMM via an LDS accumulator stash -> no gsp/bsp HBM round trip.
// - Halo tile staged with GLOBAL_LOAD_ASYNC_TO_LDS_B128 when available.
// ---------------------------------------------------------------------------

typedef __attribute__((ext_vector_type(16))) __bf16 v16bf;
typedef __attribute__((ext_vector_type(8)))  __bf16 v8bf;
typedef __attribute__((ext_vector_type(8)))  float  v8f;

#define Bn 4
#define Cc 64
#define Hh 128
#define Ww 128
#define Ff 5
#define Ll 512
#define NH 128

#if __has_builtin(__builtin_amdgcn_global_load_async_to_lds_b128)
#define HAVE_ASYNC_LDS 1
#define AS1 __attribute__((address_space(1)))
#define AS3 __attribute__((address_space(3)))
typedef __attribute__((__vector_size__(4 * sizeof(int)))) int v4i_;
#else
#define HAVE_ASYNC_LDS 0
#endif

__device__ __forceinline__ v16bf cat16(v8bf lo, v8bf hi) {
  v16bf r;
#pragma unroll
  for (int i = 0; i < 8; i++) { r[i] = lo[i]; r[i + 8] = hi[i]; }
  return r;
}

// --- instance-norm statistics: one block per (b,c) plane --------------------
__global__ void k_stats(const float* __restrict__ x, float* __restrict__ mean,
                        float* __restrict__ rstd) {
  int p = blockIdx.x;                       // 0..255
  const float* px = x + (size_t)p * (Hh * Ww);
  float s = 0.f, s2 = 0.f;
  for (int i = threadIdx.x; i < Hh * Ww; i += 256) {
    float v = px[i]; s += v; s2 += v * v;
  }
  __shared__ float rs[256], rs2[256];
  rs[threadIdx.x] = s; rs2[threadIdx.x] = s2; __syncthreads();
  for (int o = 128; o > 0; o >>= 1) {
    if (threadIdx.x < o) { rs[threadIdx.x] += rs[threadIdx.x + o];
                           rs2[threadIdx.x] += rs2[threadIdx.x + o]; }
    __syncthreads();
  }
  if (threadIdx.x == 0) {
    float m = rs[0] * (1.f / (Hh * Ww));
    float var = rs2[0] * (1.f / (Hh * Ww)) - m * m;   // biased var (ref)
    mean[p] = m; rstd[p] = rsqrtf(var + 1e-5f);
  }
}

// --- mu[b,f,k] = relu(codes[b,f,:] . fc_w[f,k,:] + fc_b[f,k]) ---------------
__global__ void k_mu(const float* __restrict__ codes, const float* __restrict__ fcw,
                     const float* __restrict__ fcb, float* __restrict__ mu) {
  int bf = blockIdx.x;                      // 0..19
  int f = bf % Ff;
  __shared__ float sc[Ll];
  const float* cp = codes + (size_t)bf * Ll;
  for (int i = threadIdx.x; i < Ll; i += 256) sc[i] = cp[i];
  __syncthreads();
  for (int k = threadIdx.x; k < Ll; k += 256) {
    const float* wr = fcw + ((size_t)f * Ll + k) * Ll;
    float acc = fcb[f * Ll + k];
    for (int l = 0; l < Ll; l++) acc += sc[l] * wr[l];
    mu[(size_t)bf * Ll + k] = fmaxf(acc, 0.f);
  }
}

// --- region map: last j with segmap>0, else 5 (zero region) -----------------
__global__ void k_region(const float* __restrict__ seg, unsigned char* __restrict__ reg) {
  int idx = blockIdx.x * 256 + threadIdx.x;  // 0..65535
  int b = idx >> 14, hw = idx & 16383;
  int r = 5;
  for (int j = Ff - 1; j >= 0; j--) {
    if (seg[((size_t)(b * Ff + j) << 14) + hw] > 0.f) { r = j; break; }
  }
  reg[idx] = (unsigned char)r;
}

// --- per-(b,region,tap) conv contributions: T[b][r][t][c] -------------------
__global__ void k_taps(const float* __restrict__ mu, const float* __restrict__ gw,
                       const float* __restrict__ bw, float* __restrict__ Tg,
                       float* __restrict__ Tb) {
  int blk = blockIdx.x; int b = blk / 6, r = blk % 6;
  size_t base = (size_t)(b * 6 + r) * 576;
  if (r == 5) {                              // zero region -> zero table
    for (int i = threadIdx.x; i < 576; i += 256) { Tg[base + i] = 0.f; Tb[base + i] = 0.f; }
    return;
  }
  __shared__ float sm[Ll];
  const float* mp = mu + (size_t)(b * Ff + r) * Ll;
  for (int i = threadIdx.x; i < Ll; i += 256) sm[i] = mp[i];
  __syncthreads();
  for (int idx = threadIdx.x; idx < 1152; idx += 256) {
    int g = idx / 576, o = idx % 576, t = o / 64, c = o % 64;
    const float* wp = (g ? bw : gw) + (size_t)c * Ll * 9 + t;   // W[c][l][t], l stride 9
    float acc = 0.f;
    for (int l = 0; l < Ll; l++) acc += sm[l] * wp[l * 9];
    (g ? Tb : Tg)[base + o] = acc;
  }
}

// --- SPADE shared conv: mask(3ch) -> relu -> actv bf16 NHWC -----------------
__global__ void __launch_bounds__(128) k_sharedconv(
    const float* __restrict__ mask, const float* __restrict__ sw,
    const float* __restrict__ sb, __bf16* __restrict__ actv) {
  int p = blockIdx.x;                       // 0..65535 pixel id
  int b = p >> 14, hw = p & 16383, h = hw >> 7, w = hw & 127;
  __shared__ float sm[27];
  int tid = threadIdx.x;                    // output channel n (0..127)
  if (tid < 27) {
    int cin = tid / 9, t = tid % 9, dh = t / 3 - 1, dw = t % 3 - 1;
    int hh = h + dh, ww = w + dw; float v = 0.f;
    if (hh >= 0 && hh < Hh && ww >= 0 && ww < Ww)
      v = mask[((size_t)(b * 3 + cin) << 14) + (hh << 7) + ww];
    sm[tid] = v;
  }
  __syncthreads();
  const float* wr = sw + tid * 27;          // [n][cin][3][3] contiguous
  float acc = sb[tid];
#pragma unroll
  for (int j = 0; j < 27; j++) acc += sm[j] * wr[j];
  actv[((size_t)p << 7) + tid] = (__bf16)fmaxf(acc, 0.f);
}

// --- weight transform: Aws[c_total][k=t*128+cin] bf16, c_total=gamma||beta --
__global__ void k_weights(const float* __restrict__ gw, const float* __restrict__ bw,
                          __bf16* __restrict__ Aws) {
  int idx = blockIdx.x * 256 + threadIdx.x; // 0..147455
  int ct = idx / 1152, k = idx % 1152, t = k / 128, cin = k % 128;
  const float* src = (ct < 64) ? gw : bw;
  int c = (ct < 64) ? ct : ct - 64;
  Aws[idx] = (__bf16)src[((size_t)(c * 128 + cin)) * 9 + t];
}

// --- fused SPADE GEMM + epilogue --------------------------------------------
// Per wg: (b, h, 32-pixel w segment). 4 waves, each: 2 M-tiles x 2 N-tiles.
// K = tap*128 + cin, 36 steps of 32. Weights (A) streamed from L2 (b128),
// actv halo (B) from an LDS tile staged via async-to-LDS. After the K loop
// the accumulators are stashed in LDS (over the halo tile) and the full
// epilogue (norm, region tables, blend) runs in-workgroup -> direct `out`.
// LDS layout: [0,26112) halo bf16 [3][34][128]  (reused as sd f32 [128][32])
//             [26112,53760) tables f32 [2][6][9][64]
//             [53760,53888) region rows u8 [3][36]
__global__ void __launch_bounds__(128) k_spade_fused(
    const __bf16* __restrict__ actv, const __bf16* __restrict__ Aws,
    const float* __restrict__ x, const float* __restrict__ mean,
    const float* __restrict__ rstd, const float* __restrict__ Tg,
    const float* __restrict__ Tb, const unsigned char* __restrict__ reg,
    const float* __restrict__ cgb, const float* __restrict__ cbb,
    const float* __restrict__ sgb, const float* __restrict__ sbb,
    const float* __restrict__ blg, const float* __restrict__ blb,
    float* __restrict__ out) {
  __shared__ __align__(32) char smem[53888];
  __bf16*        sa   = (__bf16*)smem;                  // halo during K loop
  float*         sd   = (float*)smem;                   // acc stash after it
  float*         sT   = (float*)(smem + 26112);         // [g][r][t][c]
  unsigned char* sreg = (unsigned char*)(smem + 53760); // [3][36]

  int blk = blockIdx.x;
  int w0 = (blk & 3) << 5, h = (blk >> 2) & 127, b = blk >> 9;
  int tid = threadIdx.x;

  // stage tables (both gamma & beta, 6912 floats) as float4
  {
    const float4* tg4 = (const float4*)(Tg + (size_t)b * 3456);
    const float4* tb4 = (const float4*)(Tb + (size_t)b * 3456);
    float4* st4 = (float4*)sT;
    for (int i = tid; i < 1728; i += 128)
      st4[i] = (i < 864) ? tg4[i] : tb4[i - 864];
  }
  // stage region rows h-1..h+1, w0-1..w0+32 (pad -> zero region 5)
  for (int i = tid; i < 3 * 34; i += 128) {
    int dh = i / 34, wi = i % 34; int hh = h + dh - 1, ww = w0 + wi - 1;
    unsigned char r = 5;
    if (hh >= 0 && hh < Hh && ww >= 0 && ww < Ww)
      r = reg[((size_t)b << 14) + (hh << 7) + ww];
    sreg[dh * 36 + wi] = r;
  }
  // stage 3x34x128 bf16 halo tile (zero-padded), 16B chunks
  for (int i = tid; i < 102 * 16; i += 128) {
    int p = i >> 4, q = i & 15;
    int dh = p / 34, wl = p % 34;
    int hh = h + dh - 1, ww = w0 + wl - 1;
    char* ldst = smem + p * 256 + q * 16;
    if (hh >= 0 && hh < Hh && ww >= 0 && ww < Ww) {
      const char* gsrc = (const char*)(actv +
          ((((size_t)b * Hh + hh) * Ww + ww) << 7)) + q * 16;
#if HAVE_ASYNC_LDS
      __builtin_amdgcn_global_load_async_to_lds_b128(
          (AS1 v4i_*)gsrc, (AS3 v4i_*)ldst, 0, 0);
#else
      *(uint4*)ldst = *(const uint4*)gsrc;
#endif
    } else {
      uint4 z = {0u, 0u, 0u, 0u};
      *(uint4*)ldst = z;
    }
  }
#if HAVE_ASYNC_LDS
  asm volatile("s_wait_asynccnt 0" ::: "memory");
#endif
  __syncthreads();

  int wave = tid >> 5, lane = tid & 31, half = lane >> 4, ln = lane & 15;
  const __bf16* ap0 = Aws + (size_t)(wave * 32 + ln) * 1152;  // A row = channel
  const __bf16* ap1 = ap0 + 16 * 1152;
  v8f acc00 = {}, acc01 = {}, acc10 = {}, acc11 = {};
#pragma unroll 2
  for (int ks = 0; ks < 36; ks++) {
    int k0 = ks << 5;
    int t = ks >> 2, cin0 = (ks & 3) << 5;
    int dh = t / 3, dw = t % 3;
    // B fragments: lane = pixel column; 16 consecutive K (cin) per lane-half
    const __bf16* bp = &sa[((dh * 34) + (ln + dw)) * 128 + cin0 + half * 16];
    v16bf b0 = *(const v16bf*)bp;
    v16bf b1 = *(const v16bf*)(bp + 16 * 128);
    // A fragments: lane = channel row; K halves {0-15}/{16-31}
    v16bf a0 = cat16(*(const v8bf*)(ap0 + k0 + half * 16),
                     *(const v8bf*)(ap0 + k0 + 8 + half * 16));
    v16bf a1 = cat16(*(const v8bf*)(ap1 + k0 + half * 16),
                     *(const v8bf*)(ap1 + k0 + 8 + half * 16));
    acc00 = __builtin_amdgcn_wmma_f32_16x16x32_bf16(false, a0, false, b0,
                                                    (short)0, acc00, false, false);
    acc01 = __builtin_amdgcn_wmma_f32_16x16x32_bf16(false, a0, false, b1,
                                                    (short)0, acc01, false, false);
    acc10 = __builtin_amdgcn_wmma_f32_16x16x32_bf16(false, a1, false, b0,
                                                    (short)0, acc10, false, false);
    acc11 = __builtin_amdgcn_wmma_f32_16x16x32_bf16(false, a1, false, b1,
                                                    (short)0, acc11, false, false);
  }
  __syncthreads();          // done reading halo; reuse LDS as acc stash
  // D tile: row = r + 8*half, col = ln  ->  sd[ct][px]
#pragma unroll
  for (int r = 0; r < 8; r++) {
    int ct0 = wave * 32 + r + 8 * half;
    int ct1 = ct0 + 16;
    sd[ct0 * 32 + ln]      = acc00[r];
    sd[ct0 * 32 + 16 + ln] = acc01[r];
    sd[ct1 * 32 + ln]      = acc10[r];
    sd[ct1 * 32 + 16 + ln] = acc11[r];
  }
  __syncthreads();

  // fused epilogue over 64ch x 32px
  float ga = 1.f / (1.f + __expf(-blg[0]));
  float ba = 1.f / (1.f + __expf(-blb[0]));
  int px = tid & 31, c0 = tid >> 5;
  for (int c = c0; c < Cc; c += 4) {
    size_t pix = ((((size_t)b * Cc + c) * Hh + h) << 7) + (w0 + px);
    float xn = (x[pix] - mean[b * Cc + c]) * rstd[b * Cc + c];
    float gavg = cgb[c], bavg = cbb[c];
#pragma unroll
    for (int t = 0; t < 9; t++) {
      int dh = t / 3, dw = t % 3;
      int r = sreg[dh * 36 + px + dw];
      gavg += sT[r * 576 + t * 64 + c];
      bavg += sT[3456 + r * 576 + t * 64 + c];
    }
    float gspv = sd[c * 32 + px] + sgb[c];
    float bspv = sd[(64 + c) * 32 + px] + sbb[c];
    float gf = ga * gavg + (1.f - ga) * gspv;
    float bf = ba * bavg + (1.f - ba) * bspv;
    out[pix] = xn * (1.f + gf) + bf;
  }
}

// ---------------------------------------------------------------------------
extern "C" void kernel_launch(void* const* d_in, const int* in_sizes, int n_in,
                              void* d_out, int out_size, void* d_ws, size_t ws_size,
                              hipStream_t stream) {
  const float* x     = (const float*)d_in[0];
  const float* seg   = (const float*)d_in[1];
  const float* codes = (const float*)d_in[2];
  const float* mask  = (const float*)d_in[3];
  const float* fcw   = (const float*)d_in[4];
  const float* fcb   = (const float*)d_in[5];
  const float* cgw   = (const float*)d_in[6];
  const float* cgb   = (const float*)d_in[7];
  const float* cbw   = (const float*)d_in[8];
  const float* cbb   = (const float*)d_in[9];
  const float* ssw   = (const float*)d_in[10];
  const float* ssb   = (const float*)d_in[11];
  const float* sgw   = (const float*)d_in[12];
  const float* sgb   = (const float*)d_in[13];
  const float* sbw   = (const float*)d_in[14];
  const float* sbb   = (const float*)d_in[15];
  const float* blg   = (const float*)d_in[16];
  const float* blb   = (const float*)d_in[17];

  char* ws = (char*)d_ws;
  float*         mean = (float*)(ws + 0);          // 256 f
  float*         rstd = (float*)(ws + 1024);       // 256 f
  float*         mu   = (float*)(ws + 2048);       // 4*5*512 f
  unsigned char* reg  = (unsigned char*)(ws + 43008);   // 65536 B
  float*         Tg   = (float*)(ws + 108544);     // 4*6*9*64 f
  float*         Tb   = (float*)(ws + 163840);     // 4*6*9*64 f
  __bf16*        Aws  = (__bf16*)(ws + 219136);    // 128*1152 bf16
  __bf16*        actv = (__bf16*)(ws + 514048);    // 4*128*128*128 bf16 (NHWC)
  float*         out  = (float*)d_out;

  k_stats      <<<256,   256, 0, stream>>>(x, mean, rstd);
  k_mu         <<<20,    256, 0, stream>>>(codes, fcw, fcb, mu);
  k_region     <<<256,   256, 0, stream>>>(seg, reg);
  k_taps       <<<24,    256, 0, stream>>>(mu, cgw, cbw, Tg, Tb);
  k_sharedconv <<<65536, 128, 0, stream>>>(mask, ssw, ssb, actv);
  k_weights    <<<576,   256, 0, stream>>>(sgw, sbw, Aws);
  k_spade_fused<<<2048,  128, 0, stream>>>(actv, Aws, x, mean, rstd, Tg, Tb, reg,
                                           cgb, cbb, sgb, sbb, blg, blb, out);
}
